// FeatureAttention_84129819394674
// MI455X (gfx1250) — matrix operand order repaired and checked by reading, not compiled, not run on hardware
//
#include <hip/hip_runtime.h>

typedef __attribute__((ext_vector_type(16))) _Float16 v16h;
typedef __attribute__((ext_vector_type(8)))  _Float16 v8h;
typedef __attribute__((ext_vector_type(8)))  float    v8f;

// Problem constants (from reference)
constexpr int Bc = 4, Cc = 128, Hc = 128, Wc = 128;
constexpr int Gc = 4, Cg = 32, KT = 9, Oc = 128;
constexpr int HW = Hc * Wc;
constexpr int KC = Gc * KT;      // 36 K-chunks of 32 channels each -> K = 1152
constexpr int TILE_N = 64;       // spatial positions per block
constexpr int BS_STRIDE = 40;    // padded LDS stride in halfs (80B/row -> conflict-free)
constexpr int BS_BUF = TILE_N * BS_STRIDE;   // one ping-pong buffer (2560 halfs)

// ---------------------------------------------------------------------------
// Kernel 1: convert deform weights f32 -> f16, reorder to [o][kc][c] so that
// per-lane WMMA A-fragments are contiguous 16B chunks.
//   wh[(o*36 + g*9 + kt)*32 + c] = (f16) w_deform[o][g*32 + c][kt/3][kt%3]
// ---------------------------------------------------------------------------
__global__ void dcn_wcvt(const float* __restrict__ wd, _Float16* __restrict__ wh) {
    int idx = blockIdx.x * 256 + threadIdx.x;
    if (idx >= Oc * KC * Cg) return;
    int c  = idx & 31;
    int kc = (idx >> 5) % KC;
    int o  = idx / (KC * Cg);
    int g  = kc / KT;
    int kt = kc % KT;
    wh[idx] = (_Float16)wd[((size_t)(o * Cc + g * Cg + c)) * KT + kt];
}

// ---------------------------------------------------------------------------
// Main kernel: implicit-GEMM deformable conv.
// Block: 256 threads = 8 waves. M=128 (16 rows/wave), N=64 (4 n-tiles/wave),
// K=1152 streamed as 36 chunks of 32. Ping-pong B tile -> 1 barrier/chunk.
// ---------------------------------------------------------------------------
__global__ __launch_bounds__(256)
void dcn_main(const float* __restrict__ x, const float* __restrict__ shp,
              const float* __restrict__ wo, const _Float16* __restrict__ wh,
              float* __restrict__ out) {
    __shared__ float s_wo[KC * 2 * 2];            // 1x1 offset-conv weights (144 floats)
    __shared__ int   s_y0[KC * TILE_N];           // bilinear corner row
    __shared__ int   s_x0[KC * TILE_N];           // bilinear corner col
    __shared__ float s_wy[KC * TILE_N];           // frac y
    __shared__ float s_wx[KC * TILE_N];           // frac x
    __shared__ __align__(32) _Float16 s_bs[2 * BS_BUF];  // ping-pong B tiles [n][c]

    const int tid  = threadIdx.x;
    const int lane = tid & 31;
    const int wave = tid >> 5;                    // 0..7 -> M-tile index
    const int tiles_per_b = HW / TILE_N;          // 256
    const int b  = blockIdx.x / tiles_per_b;
    const int t0 = (blockIdx.x % tiles_per_b) * TILE_N;

    if (tid < KC * 2 * 2) s_wo[tid] = wo[tid];
    __syncthreads();

    // ---- Phase A: bilinear sampling parameters for all (g,kt,n) of this tile
    for (int i = tid; i < KC * TILE_N; i += 256) {
        int n   = i & (TILE_N - 1);
        int kc  = i / TILE_N;
        int g   = kc / KT;
        int kt  = kc % KT;
        int pos = t0 + n;
        int h   = pos / Wc;
        int w   = pos % Wc;
        float s0 = shp[(size_t)(b * 2 + 0) * HW + pos];
        float s1 = shp[(size_t)(b * 2 + 1) * HW + pos];
        int row = (g * KT + kt) * 2;              // offset-conv output channel pair
        float offy = s_wo[(row + 0) * 2 + 0] * s0 + s_wo[(row + 0) * 2 + 1] * s1;
        float offx = s_wo[(row + 1) * 2 + 0] * s0 + s_wo[(row + 1) * 2 + 1] * s1;
        float py = offy + (float)(kt / 3 - 1) + (float)h;
        float px = offx + (float)(kt % 3 - 1) + (float)w;
        float fy = floorf(py), fx = floorf(px);
        s_y0[i] = (int)fy;
        s_x0[i] = (int)fx;
        s_wy[i] = py - fy;
        s_wx[i] = px - fx;
    }
    __syncthreads();

    v8f acc0 = {}, acc1 = {}, acc2 = {}, acc3 = {};

    // Per-lane fragment addressing (CDNA5 ISA layouts)
    const int m     = wave * 16 + (lane & 15);          // A row (lanes l and l+16 share M)
    const int koff  = (lane & 16) ? 8 : 0;              // A: K quadrant offset by lane half
    const int nbase = (lane & 15) * BS_STRIDE + ((lane & 16) >> 1); // B: K=(e>>3)*16+(lane&16)/2+(e&7), n=lane&15

    // Gather assignment: thread owns one spatial n, walks 8 channels (c0 + 4j)
    const int gn  = tid & 63;
    const int gc0 = tid >> 6;                           // 0..3

    for (int kc = 0; kc < KC; ++kc) {
        const int g   = kc / KT;
        const int buf = kc & 1;
        const float* xg = x + (size_t)(b * Cc + g * Cg) * HW;

        // ---- Phase B1: branchless bilinear gather of B tile (32 ch x 64 pos)
        {
            int pi = kc * TILE_N + gn;
            int y0 = s_y0[pi], x0 = s_x0[pi];
            float wy1 = s_wy[pi], wx1 = s_wx[pi];
            float wy0 = 1.f - wy1, wx0 = 1.f - wx1;
            int y1 = y0 + 1, x1 = x0 + 1;
            // validity folded into weights; indices clamped (matches reference clip*valid)
            float vy0 = ((unsigned)y0 < (unsigned)Hc) ? 1.f : 0.f;
            float vy1 = ((unsigned)y1 < (unsigned)Hc) ? 1.f : 0.f;
            float vx0 = ((unsigned)x0 < (unsigned)Wc) ? 1.f : 0.f;
            float vx1 = ((unsigned)x1 < (unsigned)Wc) ? 1.f : 0.f;
            int cy0 = min(max(y0, 0), Hc - 1), cy1 = min(max(y1, 0), Hc - 1);
            int cx0 = min(max(x0, 0), Wc - 1), cx1 = min(max(x1, 0), Wc - 1);
            float w00 = wy0 * wx0 * vy0 * vx0;
            float w01 = wy0 * wx1 * vy0 * vx1;
            float w10 = wy1 * wx0 * vy1 * vx0;
            float w11 = wy1 * wx1 * vy1 * vx1;
            int i00 = cy0 * Wc + cx0, i01 = cy0 * Wc + cx1;
            int i10 = cy1 * Wc + cx0, i11 = cy1 * Wc + cx1;
            _Float16* bs = s_bs + buf * BS_BUF + gn * BS_STRIDE;
            const float* xc = xg + (size_t)gc0 * HW;
#pragma unroll
            for (int j = 0; j < 8; ++j) {
                float v = w00 * xc[i00] + w01 * xc[i01] + w10 * xc[i10] + w11 * xc[i11];
                bs[gc0 + j * 4] = (_Float16)v;
                xc += 4 * HW;
            }
        }
        __syncthreads();   // single barrier: build(i) -> consume(i); buffer i reused at i+2

        // ---- Phase B2: A fragment from global (f16 weights, L2-resident)
        // element e: K = (e>>3)*16 + koff + (e&7) -> two contiguous v8h loads
        const _Float16* wp = wh + (size_t)(m * KC + kc) * Cg;
        v8h alo = *(const v8h*)(wp + koff);
        v8h ahi = *(const v8h*)(wp + 16 + koff);
        v16h a;
#pragma unroll
        for (int i = 0; i < 8; ++i) { a[i] = alo[i]; a[8 + i] = ahi[i]; }

        // ---- Phase B3: B fragments from LDS (element e: K=(e>>3)*16+(lane&16)/2+(e&7))
        v16h bf0, bf1, bf2, bf3;
        {
            const _Float16* p0 = s_bs + buf * BS_BUF + nbase;
            const _Float16* p1 = p0 + 16 * BS_STRIDE;
            const _Float16* p2 = p0 + 32 * BS_STRIDE;
            const _Float16* p3 = p0 + 48 * BS_STRIDE;
            v8h l0 = *(const v8h*)p0,       h0 = *(const v8h*)(p0 + 16);
            v8h l1 = *(const v8h*)p1,       h1 = *(const v8h*)(p1 + 16);
            v8h l2 = *(const v8h*)p2,       h2 = *(const v8h*)(p2 + 16);
            v8h l3 = *(const v8h*)p3,       h3 = *(const v8h*)(p3 + 16);
#pragma unroll
            for (int i = 0; i < 8; ++i) {
                bf0[i] = l0[i]; bf0[8 + i] = h0[i];
                bf1[i] = l1[i]; bf1[8 + i] = h1[i];
                bf2[i] = l2[i]; bf2[8 + i] = h2[i];
                bf3[i] = l3[i]; bf3[8 + i] = h3[i];
            }
        }

        acc0 = __builtin_amdgcn_wmma_f32_16x16x32_f16(false, a, false, bf0, (short)0, acc0, false, false);
        acc1 = __builtin_amdgcn_wmma_f32_16x16x32_f16(false, a, false, bf1, (short)0, acc1, false, false);
        acc2 = __builtin_amdgcn_wmma_f32_16x16x32_f16(false, a, false, bf2, (short)0, acc2, false, false);
        acc3 = __builtin_amdgcn_wmma_f32_16x16x32_f16(false, a, false, bf3, (short)0, acc3, false, false);
    }

    // ---- Phase C: epilogue, f32 C layout: VGPR r -> M = r (lanes 0-15) / 8+r (16-31)
    const int orow0 = wave * 16 + ((lane >> 4) & 1) * 8;
    const int ncol  = lane & 15;
#pragma unroll
    for (int r = 0; r < 8; ++r) {
        int o = orow0 + r;
        float* orow = out + (size_t)(b * Oc + o) * HW + t0;
        orow[ncol +  0] = fmaxf(acc0[r], 0.f);
        orow[ncol + 16] = fmaxf(acc1[r], 0.f);
        orow[ncol + 32] = fmaxf(acc2[r], 0.f);
        orow[ncol + 48] = fmaxf(acc3[r], 0.f);
    }
}

// ---------------------------------------------------------------------------
extern "C" void kernel_launch(void* const* d_in, const int* in_sizes, int n_in,
                              void* d_out, int out_size, void* d_ws, size_t ws_size,
                              hipStream_t stream) {
    const float* x   = (const float*)d_in[0];   // [4,128,128,128]
    const float* shp = (const float*)d_in[1];   // [4,2,128,128]
    const float* wo  = (const float*)d_in[2];   // [72,2]
    const float* wd  = (const float*)d_in[3];   // [128,128,3,3]
    float* outp = (float*)d_out;
    _Float16* wh = (_Float16*)d_ws;             // 147456 halfs = 288KB scratch

    int nw = Oc * KC * Cg;                      // 147456
    dcn_wcvt<<<(nw + 255) / 256, 256, 0, stream>>>(wd, wh);

    int blocks = Bc * (HW / TILE_N);            // 1024
    dcn_main<<<blocks, 256, 0, stream>>>(x, shp, wo, wh, outp);
}
